// MultiHeadAttention_35802847379691
// MI455X (gfx1250) — matrix operand order repaired and checked by reading, not compiled, not run on hardware
//
#include <hip/hip_runtime.h>

// ---------- types ----------
typedef __attribute__((ext_vector_type(16))) __bf16 bf16x16;
typedef __attribute__((ext_vector_type(8)))  float  f32x8;
typedef __attribute__((ext_vector_type(4)))  unsigned int u32x4;
typedef __attribute__((ext_vector_type(4)))  int i32x4;

union Frag {
    bf16x16 v;
    u32x4   u[2];
};

__device__ __forceinline__ f32x8 zero8() {
    f32x8 z;
#pragma unroll
    for (int e = 0; e < 8; ++e) z[e] = 0.0f;
    return z;
}

__device__ __forceinline__ f32x8 wmma_bf16(const bf16x16& a, const bf16x16& b, const f32x8& c) {
    return __builtin_amdgcn_wmma_f32_16x16x32_bf16(false, a, false, b, (short)0, c, false, false);
}

// ---------- 16-lane row max reduction via DPP (stays in VALU, co-executes) ----------
#if __has_builtin(__builtin_amdgcn_update_dpp)
template <int CTRL>
__device__ __forceinline__ float dppmov(float x) {
    int xi = __builtin_bit_cast(int, x);
    int r = __builtin_amdgcn_update_dpp(0, xi, CTRL, 0xF, 0xF, true);
    return __builtin_bit_cast(float, r);
}
__device__ __forceinline__ float redmax16(float x) {
    x = fmaxf(x, dppmov<0xB1>(x));   // quad_perm [1,0,3,2]  (xor 1)
    x = fmaxf(x, dppmov<0x4E>(x));   // quad_perm [2,3,0,1]  (xor 2)
    x = fmaxf(x, dppmov<0x141>(x));  // row_half_mirror      (xor 7 within 8)
    x = fmaxf(x, dppmov<0x140>(x));  // row_mirror           (xor 15 within 16)
    return x;
}
#else
__device__ __forceinline__ float redmax16(float x) {
    x = fmaxf(x, __shfl_xor(x, 1, 32));
    x = fmaxf(x, __shfl_xor(x, 2, 32));
    x = fmaxf(x, __shfl_xor(x, 4, 32));
    x = fmaxf(x, __shfl_xor(x, 8, 32));
    return x;
}
#endif

// ---------- async global -> LDS copy (ASYNCcnt path), with sync fallback ----------
#if __has_builtin(__builtin_amdgcn_global_load_async_to_lds_b128)
#define HAVE_ASYNC_LDS 1
typedef __attribute__((address_space(1))) i32x4* gptr_i32x4;
typedef __attribute__((address_space(3))) i32x4* lptr_i32x4;
#endif

__device__ __forceinline__ void async_cp16(const void* g, void* l) {
#ifdef HAVE_ASYNC_LDS
    // Generic pointer bits: full 64 = global address; low 32 = LDS byte offset.
    gptr_i32x4 gp = (gptr_i32x4)(unsigned long long)(__SIZE_TYPE__)g;
    lptr_i32x4 lp = (lptr_i32x4)(unsigned int)(__SIZE_TYPE__)l;
    __builtin_amdgcn_global_load_async_to_lds_b128(gp, lp, 0, 0);
#else
    *(u32x4*)l = *(const u32x4*)g;
#endif
}

__device__ __forceinline__ void wait_async() {
#ifdef HAVE_ASYNC_LDS
#if __has_builtin(__builtin_amdgcn_s_wait_asynccnt)
    __builtin_amdgcn_s_wait_asynccnt(0);
#endif
#endif
}

// ---------- fp32 -> bf16 convert ----------
__global__ void cvt_f32_bf16(const float* __restrict__ s, __bf16* __restrict__ d, int n) {
    int i = blockIdx.x * blockDim.x + threadIdx.x;
    if (i < n) d[i] = (__bf16)s[i];
}

// ---------- WMMA GEMM: Y = (A @ W^T + bias) * scale ----------
// MODE 0: bf16 -> head layout [B,NH,S,DK]   (Q, K projections)
// MODE 1: bf16 -> transposed  [B,NH,DK,S]   (V projection)
// MODE 2: fp32 -> [M,N]                     (output projection)
template <int MODE>
__global__ __launch_bounds__(256) void gemm_bf16_wmma(
    const __bf16* __restrict__ A, const __bf16* __restrict__ W,
    const float* __restrict__ bias, void* __restrict__ Yv,
    int Mdim, int N, int K, float scale)
{
    const int lane = threadIdx.x & 31;
    const int wave = threadIdx.x >> 5;
    const int lrow = lane & 15;
    const int half = lane >> 4;
    const int mBase = blockIdx.x * 128 + (wave >> 2) * 64;
    const int nBase = blockIdx.y * 128 + (wave & 3) * 32;

    f32x8 acc[4][2];
#pragma unroll
    for (int i = 0; i < 4; ++i)
#pragma unroll
        for (int j = 0; j < 2; ++j) acc[i][j] = zero8();

    const __bf16* aRow[4];
    const __bf16* wRow[2];
#pragma unroll
    for (int i = 0; i < 4; ++i) aRow[i] = A + (size_t)(mBase + i * 16 + lrow) * K;
#pragma unroll
    for (int j = 0; j < 2; ++j) wRow[j] = W + (size_t)(nBase + j * 16 + lrow) * K;

#pragma unroll 2
    for (int k0 = 0; k0 < K; k0 += 32) {
        Frag a[4], b[2];
#pragma unroll
        for (int i = 0; i < 4; ++i) {
            const __bf16* p = aRow[i] + k0 + half * 16;  // A: K = half*16 .. +15
            a[i].u[0] = *(const u32x4*)p;
            a[i].u[1] = *(const u32x4*)(p + 8);
        }
#pragma unroll
        for (int j = 0; j < 2; ++j) {
            const __bf16* p = wRow[j] + k0 + half * 16;  // B: K = half*16 .. +15
            b[j].u[0] = *(const u32x4*)p;
            b[j].u[1] = *(const u32x4*)(p + 8);
        }
#pragma unroll
        for (int i = 0; i < 4; ++i)
#pragma unroll
            for (int j = 0; j < 2; ++j)
                acc[i][j] = wmma_bf16(a[i].v, b[j].v, acc[i][j]);
    }

#pragma unroll
    for (int j = 0; j < 2; ++j) {
        const int n = nBase + j * 16 + lrow;
        const float bv = bias[n];
#pragma unroll
        for (int i = 0; i < 4; ++i) {
#pragma unroll
            for (int v = 0; v < 8; ++v) {
                const int m = mBase + i * 16 + v + 8 * half;
                const float val = (acc[i][j][v] + bv) * scale;
                if (MODE == 2) {
                    ((float*)Yv)[(size_t)m * N + n] = val;
                } else {
                    const int bb = m >> 11, s = m & 2047;   // S = 2048
                    const int h = n >> 6, d = n & 63;       // DK = 64
                    const size_t idx = (MODE == 0)
                        ? ((size_t)((bb * 16 + h) * 2048 + s)) * 64 + d
                        : ((size_t)((bb * 16 + h) * 64 + d)) * 2048 + s;
                    ((__bf16*)Yv)[idx] = (__bf16)val;
                }
            }
        }
    }
}

// ---------- Flash attention (64-key tiles) ----------
// Q,K: [B*NH, S, DK] bf16 (Q pre-scaled by log2(e)/sqrt(DK));  V: [B*NH, DK, S] bf16
// O: [B*S, H] bf16 with column = h*DK + d.  Softmax runs in exp2 domain.
// Row sums of P are computed on the matrix pipe via an all-ones B fragment.
__global__ __launch_bounds__(256) void flash_attn_wmma(
    const __bf16* __restrict__ Q, const __bf16* __restrict__ Kh,
    const __bf16* __restrict__ Vt, __bf16* __restrict__ O)
{
    // Double-buffered tiles: per buf, K tile [64 keys][64 d] then V tile [64 d][64 keys]
    __shared__ __align__(16) __bf16 kv[2][8192];          // 32 KB
    __shared__ __align__(16) __bf16 plds[8][16][72];      // P transpose, padded stride (144B rows)

    const int tid  = threadIdx.x;
    const int lane = tid & 31;
    const int wave = tid >> 5;
    const int lrow = lane & 15;
    const int half = lane >> 4;
    const int bh = blockIdx.y;              // b*NH + h
    const int b = bh >> 4, h = bh & 15;
    const int q0 = blockIdx.x * 128 + wave * 16;

    const __bf16* Kbase = Kh + (size_t)bh * 2048 * 64;
    const __bf16* Vbase = Vt + (size_t)bh * 64 * 2048;

    // Q A-fragments (16 rows x 64, two K=32 steps)
    Frag qa[2];
    {
        const __bf16* qRow = Q + ((size_t)bh * 2048 + q0 + lrow) * 64;
#pragma unroll
        for (int j = 0; j < 2; ++j) {
            const __bf16* p = qRow + j * 32 + half * 16;
            qa[j].u[0] = *(const u32x4*)p;
            qa[j].u[1] = *(const u32x4*)(p + 8);
        }
    }

    // All-ones B fragment: wmma(P, ones) -> per-row sums of P in C layout
    Frag ones;
#pragma unroll
    for (int e = 0; e < 16; ++e) ones.v[e] = (__bf16)1.0f;

    // cooperative tile stage: K tile = contiguous 8KB; V tile = 64 rows x 128B
    const int vd = tid >> 2, vp = tid & 3;
    auto stage = [&](int bi, int t) {
#pragma unroll
        for (int c = 0; c < 2; ++c) {
            async_cp16(Kbase + (size_t)t * 64 + c * 2048 + tid * 8, &kv[bi][c * 2048 + tid * 8]);
            async_cp16(Vbase + (size_t)vd * 2048 + t + c * 32 + vp * 8,
                       &kv[bi][4096 + vd * 64 + c * 32 + vp * 8]);
        }
    };

    float mrow[8];
    f32x8 o[4], osum;
#pragma unroll
    for (int v = 0; v < 8; ++v) mrow[v] = -INFINITY;
#pragma unroll
    for (int jd = 0; jd < 4; ++jd) o[jd] = zero8();
    osum = zero8();

    stage(0, 0);

    for (int it = 0; it < 32; ++it) {
        const int bi = it & 1;
        wait_async();
        __syncthreads();                     // tile[bi] visible to all waves
        if (it + 1 < 32) stage(bi ^ 1, (it + 1) * 64);   // prefetch next tile

        const __bf16* kt = &kv[bi][0];
        const __bf16* vt = &kv[bi][4096];

        // ---- scores: four 16x16 C tiles covering 64 keys (K frags from LDS) ----
        f32x8 s[4];
#pragma unroll
        for (int g = 0; g < 4; ++g) {
            s[g] = zero8();
#pragma unroll
            for (int j = 0; j < 2; ++j) {
                Frag kb;
                const __bf16* p = kt + (g * 16 + lrow) * 64 + j * 32 + half * 16;
                kb.u[0] = *(const u32x4*)p;
                kb.u[1] = *(const u32x4*)(p + 8);
                s[g] = wmma_bf16(qa[j].v, kb.v, s[g]);
            }
        }

        // ---- online softmax in exp2 domain (one DPP butterfly per 64 keys) ----
        float mnew[8], alpha[8], p[4][8];
#pragma unroll
        for (int v = 0; v < 8; ++v) {
            float mx = fmaxf(fmaxf(s[0][v], s[1][v]), fmaxf(s[2][v], s[3][v]));
            mx = redmax16(mx);
            mnew[v] = fmaxf(mrow[v], mx);
            alpha[v] = exp2f(mrow[v] - mnew[v]);
            mrow[v] = mnew[v];
#pragma unroll
            for (int g = 0; g < 4; ++g) p[g][v] = exp2f(s[g][v] - mnew[v]);
        }
#pragma unroll
        for (int jd = 0; jd < 4; ++jd)
#pragma unroll
            for (int v = 0; v < 8; ++v) o[jd][v] *= alpha[v];
#pragma unroll
        for (int v = 0; v < 8; ++v) osum[v] *= alpha[v];

        // ---- P: C layout -> A layout via LDS ----
#pragma unroll
        for (int g = 0; g < 4; ++g)
#pragma unroll
            for (int v = 0; v < 8; ++v)
                plds[wave][v + 8 * half][g * 16 + lrow] = (__bf16)p[g][v];
        __syncthreads();
        Frag pa[2];
#pragma unroll
        for (int c = 0; c < 2; ++c) {
            const __bf16* rp = &plds[wave][lrow][c * 32 + half * 16];
            pa[c].u[0] = *(const u32x4*)rp;
            pa[c].u[1] = *(const u32x4*)(rp + 8);
        }
        __syncthreads();

        // ---- O += P @ V ; rowsum += P @ ones (all on the matrix pipe) ----
#pragma unroll
        for (int jd = 0; jd < 4; ++jd) {
#pragma unroll
            for (int c = 0; c < 2; ++c) {
                const __bf16* vp2 = vt + (jd * 16 + lrow) * 64 + c * 32 + half * 16;
                Frag vb;
                vb.u[0] = *(const u32x4*)vp2;
                vb.u[1] = *(const u32x4*)(vp2 + 8);
                o[jd] = wmma_bf16(pa[c].v, vb.v, o[jd]);
            }
        }
        osum = wmma_bf16(pa[0].v, ones.v, osum);
        osum = wmma_bf16(pa[1].v, ones.v, osum);
    }

    // ---- normalize + store ----
    float inv[8];
#pragma unroll
    for (int v = 0; v < 8; ++v) inv[v] = (osum[v] > 0.0f) ? (1.0f / osum[v]) : 0.0f;
#pragma unroll
    for (int jd = 0; jd < 4; ++jd) {
#pragma unroll
        for (int v = 0; v < 8; ++v) {
            const int m = b * 2048 + q0 + v + 8 * half;
            const int col = h * 64 + jd * 16 + lrow;
            O[(size_t)m * 1024 + col] = (__bf16)(o[jd][v] * inv[v]);
        }
    }
}

// ---------- launch ----------
extern "C" void kernel_launch(void* const* d_in, const int* in_sizes, int n_in,
                              void* d_out, int out_size, void* d_ws, size_t ws_size,
                              hipStream_t stream) {
    const float* x  = (const float*)d_in[0];
    const float* Wq = (const float*)d_in[1];
    const float* bq = (const float*)d_in[2];
    const float* Wk = (const float*)d_in[3];
    const float* bk = (const float*)d_in[4];
    const float* Wv = (const float*)d_in[5];
    const float* bv = (const float*)d_in[6];
    const float* Wo = (const float*)d_in[7];
    const float* bo = (const float*)d_in[8];
    float* out = (float*)d_out;

    char* ws = (char*)d_ws;
    __bf16* xb  = (__bf16*)(ws);                       //  8 MB : x bf16 [4096,1024]
    __bf16* wqb = (__bf16*)(ws + ( 8ull << 20));       //  2 MB
    __bf16* wkb = (__bf16*)(ws + (10ull << 20));       //  2 MB
    __bf16* wvb = (__bf16*)(ws + (12ull << 20));       //  2 MB
    __bf16* wob = (__bf16*)(ws + (14ull << 20));       //  2 MB
    __bf16* Qh  = (__bf16*)(ws + (16ull << 20));       //  8 MB [B,NH,S,DK]
    __bf16* Kh  = (__bf16*)(ws + (24ull << 20));       //  8 MB [B,NH,S,DK]
    __bf16* Vt  = (__bf16*)(ws + (32ull << 20));       //  8 MB [B,NH,DK,S]
    __bf16* AO  = (__bf16*)(ws + (40ull << 20));       //  8 MB attn out [4096,1024]

    const int NX = 4096 * 1024;
    const int NW = 1024 * 1024;
    cvt_f32_bf16<<<NX / 256, 256, 0, stream>>>(x,  xb,  NX);
    cvt_f32_bf16<<<NW / 256, 256, 0, stream>>>(Wq, wqb, NW);
    cvt_f32_bf16<<<NW / 256, 256, 0, stream>>>(Wk, wkb, NW);
    cvt_f32_bf16<<<NW / 256, 256, 0, stream>>>(Wv, wvb, NW);
    cvt_f32_bf16<<<NW / 256, 256, 0, stream>>>(Wo, wob, NW);

    // Q is pre-scaled by 1/sqrt(DK) * log2(e) so softmax runs natively in exp2.
    const float qscale = 0.125f * 1.44269504088896340736f;

    dim3 gg(32, 8), blk(256);
    gemm_bf16_wmma<0><<<gg, blk, 0, stream>>>(xb, wqb, bq, Qh, 4096, 1024, 1024, qscale);
    gemm_bf16_wmma<0><<<gg, blk, 0, stream>>>(xb, wkb, bk, Kh, 4096, 1024, 1024, 1.0f);
    gemm_bf16_wmma<1><<<gg, blk, 0, stream>>>(xb, wvb, bv, Vt, 4096, 1024, 1024, 1.0f);

    flash_attn_wmma<<<dim3(16, 32), blk, 0, stream>>>(Qh, Kh, Vt, AO);

    gemm_bf16_wmma<2><<<gg, blk, 0, stream>>>(AO, wob, bo, out, 4096, 1024, 1024, 1.0f);
}
